// MultiHeadAttention_34359738945
// MI455X (gfx1250) — matrix operand order repaired and checked by reading, not compiled, not run on hardware
//
#include <hip/hip_runtime.h>

// ---------------------------------------------------------------------------
// MI455X (gfx1250) fused multi-head attention, bf16 WMMA + f32 accumulate.
//   B=2, S=2048, D=1024, H=16, HD=64
// Pipeline:
//   [0] f32 -> bf16 conversion of x and the four weight matrices (packed)
//   [1] QKV projection: wave computes 32x64 tile (2 row tiles x 1 head's
//       cols), B-fragments reused across both row tiles. Q,K stored
//       [b,h,s,hd]; V stored transposed [b,h,hd,s].
//   [2] Flash attention per head: 16-query tile per wave, 32-key blocks,
//       online softmax in f32, P routed through aligned LDS tile.
//   [3] Output projection + bias (32x64 per wave), f32 output.
// All GEMMs use v_wmma_f32_16x16x32_bf16.
// ---------------------------------------------------------------------------

typedef __bf16 bf16;
typedef __attribute__((ext_vector_type(16))) __bf16 v16bf;
typedef __attribute__((ext_vector_type(8)))  __bf16 v8bf;
typedef __attribute__((ext_vector_type(8)))  float  v8f;

#define DEV static __device__ __forceinline__

static constexpr int Bc  = 2;
static constexpr int Sc  = 2048;
static constexpr int Dc  = 1024;
static constexpr int Hc  = 16;
static constexpr int HDc = 64;

union FragA { v16bf v; v8bf h[2]; };

// A-operand fragment (16x32 bf16): lane m holds row m with K chunks
// {0..7,16..23} (lanes 0-15) / {8..15,24..31} (lanes 16-31).
// Caller passes p = row_base + k + (lane>>4)*8; reads p[0..7], p[16..23].
DEV v16bf load_a(const bf16* p) {
  FragA f;
  f.h[0] = *(const v8bf*)(p);
  f.h[1] = *(const v8bf*)(p + 16);
  return f.v;
}

// B-operand fragment (32x16 bf16): lane n holds column n with K chunks
// {0..7,16..23} (lanes 0-15) / {8..15,24..31} (lanes 16-31).
// Caller passes p = col_base + k + (lane>>4)*8; reads p[0..7], p[16..23].
DEV v16bf load_b(const bf16* p) {
  FragA f;
  f.h[0] = *(const v8bf*)(p);
  f.h[1] = *(const v8bf*)(p + 16);
  return f.v;
}

DEV v8f wmma_bf16(v16bf a, v16bf b, v8f c) {
  return __builtin_amdgcn_wmma_f32_16x16x32_bf16(
      /*neg_a=*/false, a, /*neg_b=*/false, b,
      /*c_mod=*/(short)0, c, /*reuse_a=*/false, /*reuse_b=*/false);
}

// Butterfly reductions across each 16-lane half (rows live per-half in the
// WMMA C layout); xor masks <= 8 never cross the half boundary.
DEV float rowmax16(float x) {
  x = fmaxf(x, __shfl_xor(x, 1, 32));
  x = fmaxf(x, __shfl_xor(x, 2, 32));
  x = fmaxf(x, __shfl_xor(x, 4, 32));
  x = fmaxf(x, __shfl_xor(x, 8, 32));
  return x;
}
DEV float rowsum16(float x) {
  x += __shfl_xor(x, 1, 32);
  x += __shfl_xor(x, 2, 32);
  x += __shfl_xor(x, 4, 32);
  x += __shfl_xor(x, 8, 32);
  return x;
}

// ---------------------------------------------------------------------------
// [0] f32 -> bf16, two elements per thread packed into one dword store.
// ---------------------------------------------------------------------------
__global__ void cvt_bf16_kernel(const float2* __restrict__ in,
                                unsigned int* __restrict__ out, int n2) {
  int i = blockIdx.x * blockDim.x + threadIdx.x;
  int stride = gridDim.x * blockDim.x;
  for (; i < n2; i += stride) {
    float2 v = in[i];
    union { bf16 h[2]; unsigned int u; } pk;
    pk.h[0] = (bf16)v.x;
    pk.h[1] = (bf16)v.y;
    out[i] = pk.u;
  }
}

// ---------------------------------------------------------------------------
// [1] QKV projection.  grid = (16, 16, 3), block = 256 (8 waves).
// Wave computes a 32-row x 64-col tile (two 16-row tiles sharing every
// B-fragment); the 64 cols are exactly one head.
// ---------------------------------------------------------------------------
__global__ void qkv_proj_kernel(const bf16* __restrict__ xb,
                                const bf16* __restrict__ wq,
                                const bf16* __restrict__ wk,
                                const bf16* __restrict__ wv,
                                bf16* __restrict__ Qb,
                                bf16* __restrict__ Kb,
                                bf16* __restrict__ Vt) {
  const int lane  = threadIdx.x & 31;
  const int w     = threadIdx.x >> 5;
  const int cn    = lane & 15;
  const int hi    = lane >> 4;
  const int koffA = hi * 8;
  const int koffB = hi * 8;
  const int rbase = (blockIdx.x * 8 + w) * 32;   // flat row over B*S
  const int h     = blockIdx.y;                  // head == 64-col group
  const int cbase = h * 64;
  const int z     = blockIdx.z;                  // 0=Q 1=K 2=V
  const bf16* W   = (z == 0) ? wq : (z == 1) ? wk : wv;

  v8f acc[2][4] = {{{}, {}, {}, {}}, {{}, {}, {}, {}}};
  const bf16* arow0 = xb + (size_t)(rbase + cn) * Dc;
  const bf16* arow1 = xb + (size_t)(rbase + 16 + cn) * Dc;
  const bf16* brow[4];
#pragma unroll
  for (int t = 0; t < 4; ++t)
    brow[t] = W + (size_t)(cbase + t * 16 + cn) * Dc;

  for (int k = 0; k < Dc; k += 32) {
    v16bf a0 = load_a(arow0 + k + koffA);
    v16bf a1 = load_a(arow1 + k + koffA);
#pragma unroll
    for (int t = 0; t < 4; ++t) {
      v16bf b = load_b(brow[t] + k + koffB);
      acc[0][t] = wmma_bf16(a0, b, acc[0][t]);
      acc[1][t] = wmma_bf16(a1, b, acc[1][t]);
    }
  }

#pragma unroll
  for (int rr = 0; rr < 2; ++rr) {
#pragma unroll
    for (int t = 0; t < 4; ++t) {
#pragma unroll
      for (int j = 0; j < 8; ++j) {
        const int r  = rbase + rr * 16 + j + hi * 8;  // flat row (b*S + s)
        const int b  = r >> 11;                       // S == 2048
        const int s  = r & (Sc - 1);
        const int hd = t * 16 + cn;
        const bf16 val = (bf16)acc[rr][t][j];
        if (z == 2) {
          Vt[(((size_t)(b * Hc + h)) * HDc + hd) * Sc + s] = val;
        } else {
          bf16* dst = (z == 0) ? Qb : Kb;
          dst[(((size_t)(b * Hc + h)) * Sc + s) * HDc + hd] = val;
        }
      }
    }
  }
}

// ---------------------------------------------------------------------------
// [2] Flash attention.  grid = (S/64, B*H), block = 128 (4 waves).
// Each wave owns 16 query rows; iterates 32-key blocks up to the causal
// boundary with online softmax.  Per block: 4 WMMAs for scores (K=64),
// LDS bounce of P, 4 WMMAs for P@V (HD=64).
// ---------------------------------------------------------------------------
__global__ void attn_kernel(const bf16* __restrict__ Qb,
                            const bf16* __restrict__ Kb,
                            const bf16* __restrict__ Vt,
                            bf16* __restrict__ ctx) {
  // per-wave 16x32 P tile; row stride 48 elems (96 B) keeps every 16-byte
  // A-fragment chunk aligned; explicit 16-byte base alignment for b128 LDS ops
  __shared__ __align__(16) bf16 pbuf[4][16][48];

  const int lane  = threadIdx.x & 31;
  const int w     = threadIdx.x >> 5;
  const int cn    = lane & 15;
  const int hi    = lane >> 4;
  const int koffA = hi * 8;
  const int koffB = hi * 8;
  const int qbase = (blockIdx.x * 4 + w) * 16;
  const int bh    = blockIdx.y;
  const int b     = bh >> 4;         // H == 16
  const int h     = bh & (Hc - 1);

  const bf16* Qh = Qb + (size_t)bh * Sc * HDc;
  const bf16* Kh = Kb + (size_t)bh * Sc * HDc;
  const bf16* Vh = Vt + (size_t)bh * HDc * Sc;

  // Q fragments for K-steps 0 and 32 (loop invariant).
  const v16bf aq0 = load_a(Qh + (size_t)(qbase + cn) * HDc + 0  + koffA);
  const v16bf aq1 = load_a(Qh + (size_t)(qbase + cn) * HDc + 32 + koffA);

  v8f o[4] = {{}, {}, {}, {}};
  float m[8], l[8];
#pragma unroll
  for (int j = 0; j < 8; ++j) { m[j] = -3.0e38f; l[j] = 0.0f; }

  const int nb = (qbase + 47) >> 5;       // 32-key blocks covering keys<=q
  const float scale = 0.125f;             // 1/sqrt(64)

  for (int kb = 0; kb < nb; ++kb) {
    const int k0 = kb * 32;

    if (kb + 1 < nb) {   // hint next block's K/V into cache
      __builtin_prefetch(Kh + (size_t)(k0 + 32 + cn) * HDc, 0, 1);
      __builtin_prefetch(Vh + (size_t)cn * Sc + k0 + 32, 0, 1);
    }

    // scores: S[16x32] = Q[16x64] @ K^T[64x32]
    v8f s0 = {}, s1 = {};
    {
      v16bf b00 = load_b(Kh + (size_t)(k0 + cn) * HDc + 0  + koffB);
      v16bf b01 = load_b(Kh + (size_t)(k0 + cn) * HDc + 32 + koffB);
      s0 = wmma_bf16(aq0, b00, s0);
      s0 = wmma_bf16(aq1, b01, s0);
      v16bf b10 = load_b(Kh + (size_t)(k0 + 16 + cn) * HDc + 0  + koffB);
      v16bf b11 = load_b(Kh + (size_t)(k0 + 16 + cn) * HDc + 32 + koffB);
      s1 = wmma_bf16(aq0, b10, s1);
      s1 = wmma_bf16(aq1, b11, s1);
    }

    const int kc0 = k0 + cn;        // absolute key of sub-tile 0 column
    const int kc1 = k0 + 16 + cn;   // absolute key of sub-tile 1 column
#pragma unroll
    for (int j = 0; j < 8; ++j) {
      const int qi = qbase + j + hi * 8;
      float p0 = s0[j] * scale;
      float p1 = s1[j] * scale;
      if (kc0 > qi) p0 = -1.0e30f;  // causal mask
      if (kc1 > qi) p1 = -1.0e30f;
      const float rm = rowmax16(fmaxf(p0, p1));
      const float mn = fmaxf(m[j], rm);
      const float alpha = __expf(m[j] - mn);
      m[j] = mn;
      p0 = __expf(p0 - mn);
      p1 = __expf(p1 - mn);
      l[j] = l[j] * alpha + rowsum16(p0 + p1);
#pragma unroll
      for (int t = 0; t < 4; ++t) o[t][j] *= alpha;
      const int r = j + hi * 8;     // row within tile (C-layout)
      pbuf[w][r][cn]      = (bf16)p0;
      pbuf[w][r][16 + cn] = (bf16)p1;
    }

    // C-layout -> A-layout via per-wave LDS tile (wave-synchronous; LDS ops
    // from one wave are in-order, wait drains DScnt before the reads).
    asm volatile("s_wait_dscnt 0" ::: "memory");
    const v16bf pa = load_a(&pbuf[w][cn][koffA]);

    // O[16x64] += P[16x32] @ V[32x64]   (V stored transposed -> contiguous)
#pragma unroll
    for (int t = 0; t < 4; ++t) {
      v16bf vb = load_b(Vh + (size_t)(t * 16 + cn) * Sc + k0 + koffB);
      o[t] = wmma_bf16(pa, vb, o[t]);
    }
    asm volatile("s_wait_dscnt 0" ::: "memory");
  }

  // normalize and write ctx[b, s, h*64 + hd] as bf16
#pragma unroll
  for (int j = 0; j < 8; ++j) {
    const float inv = 1.0f / l[j];
    const int s = qbase + j + hi * 8;
#pragma unroll
    for (int t = 0; t < 4; ++t) {
      ctx[((size_t)(b * Sc + s)) * Dc + h * 64 + t * 16 + cn] =
          (bf16)(o[t][j] * inv);
    }
  }
}

// ---------------------------------------------------------------------------
// [3] Output projection + bias.  grid = (16, 16), block = 256 (8 waves).
// Wave computes 32x64 (two row tiles share every B-fragment).
// ---------------------------------------------------------------------------
__global__ void out_proj_kernel(const bf16* __restrict__ ctx,
                                const bf16* __restrict__ wo,
                                const float* __restrict__ bo,
                                float* __restrict__ out) {
  const int lane  = threadIdx.x & 31;
  const int w     = threadIdx.x >> 5;
  const int cn    = lane & 15;
  const int hi    = lane >> 4;
  const int koffA = hi * 8;
  const int koffB = hi * 8;
  const int rbase = (blockIdx.x * 8 + w) * 32;
  const int cbase = blockIdx.y * 64;

  v8f acc[2][4] = {{{}, {}, {}, {}}, {{}, {}, {}, {}}};
  const bf16* arow0 = ctx + (size_t)(rbase + cn) * Dc;
  const bf16* arow1 = ctx + (size_t)(rbase + 16 + cn) * Dc;
  const bf16* brow[4];
#pragma unroll
  for (int t = 0; t < 4; ++t)
    brow[t] = wo + (size_t)(cbase + t * 16 + cn) * Dc;

  for (int k = 0; k < Dc; k += 32) {
    v16bf a0 = load_a(arow0 + k + koffA);
    v16bf a1 = load_a(arow1 + k + koffA);
#pragma unroll
    for (int t = 0; t < 4; ++t) {
      v16bf b = load_b(brow[t] + k + koffB);
      acc[0][t] = wmma_bf16(a0, b, acc[0][t]);
      acc[1][t] = wmma_bf16(a1, b, acc[1][t]);
    }
  }

#pragma unroll
  for (int t = 0; t < 4; ++t) {
    const int col = cbase + t * 16 + cn;
    const float bias = bo[col];
#pragma unroll
    for (int rr = 0; rr < 2; ++rr) {
#pragma unroll
      for (int j = 0; j < 8; ++j) {
        out[(size_t)(rbase + rr * 16 + j + hi * 8) * Dc + col] =
            acc[rr][t][j] + bias;
      }
    }
  }
}

// ---------------------------------------------------------------------------
// Host-side launcher
// ---------------------------------------------------------------------------
extern "C" void kernel_launch(void* const* d_in, const int* in_sizes, int n_in,
                              void* d_out, int out_size, void* d_ws,
                              size_t ws_size, hipStream_t stream) {
  const float* x  = (const float*)d_in[0];
  const float* Wq = (const float*)d_in[1];
  const float* Wk = (const float*)d_in[2];
  const float* Wv = (const float*)d_in[3];
  const float* Wo = (const float*)d_in[4];
  const float* bo = (const float*)d_in[5];
  float* out = (float*)d_out;

  const size_t nx = (size_t)Bc * Sc * Dc;   // 4,194,304 elements
  const size_t nw = (size_t)Dc * Dc;        // 1,048,576 elements

  char* ws = (char*)d_ws;
  size_t off = 0;
  auto carve = [&](size_t bytes) -> char* {
    char* p = ws + off;
    off += (bytes + 255) & ~(size_t)255;
    return p;
  };
  bf16* xb  = (bf16*)carve(nx * 2);
  bf16* wqb = (bf16*)carve(nw * 2);
  bf16* wkb = (bf16*)carve(nw * 2);
  bf16* wvb = (bf16*)carve(nw * 2);
  bf16* wob = (bf16*)carve(nw * 2);
  bf16* Qb  = (bf16*)carve(nx * 2);
  bf16* Kb  = (bf16*)carve(nx * 2);
  bf16* Vt  = (bf16*)carve(nx * 2);
  bf16* ctx = (bf16*)carve(nx * 2);
  (void)ws_size; (void)in_sizes; (void)n_in; (void)out_size;

  cvt_bf16_kernel<<<2048, 256, 0, stream>>>((const float2*)x,
                                            (unsigned int*)xb, (int)(nx / 2));
  cvt_bf16_kernel<<<512, 256, 0, stream>>>((const float2*)Wq,
                                           (unsigned int*)wqb, (int)(nw / 2));
  cvt_bf16_kernel<<<512, 256, 0, stream>>>((const float2*)Wk,
                                           (unsigned int*)wkb, (int)(nw / 2));
  cvt_bf16_kernel<<<512, 256, 0, stream>>>((const float2*)Wv,
                                           (unsigned int*)wvb, (int)(nw / 2));
  cvt_bf16_kernel<<<512, 256, 0, stream>>>((const float2*)Wo,
                                           (unsigned int*)wob, (int)(nw / 2));

  // (B*S)/32 = 128 row tiles -> 16 blocks x 8 waves; 16 heads; z = Q/K/V
  qkv_proj_kernel<<<dim3(16, 16, 3), 256, 0, stream>>>(xb, wqb, wkb, wvb,
                                                       Qb, Kb, Vt);

  // S/16 = 128 query tiles -> 32 blocks x 4 waves; B*H = 32 head-batches
  attn_kernel<<<dim3(32, 32), 128, 0, stream>>>(Qb, Kb, Vt, ctx);

  out_proj_kernel<<<dim3(16, 16), 256, 0, stream>>>(ctx, wob, bo, out);
}
